// Attention_2491081032032
// MI455X (gfx1250) — compile-verified, matrix operand order checked by reading, not run on hardware
//
#include <hip/hip_runtime.h>

// ---------------------------------------------------------------------------
// Types for CDNA5 WMMA (wave32, 16x16x32 bf16 -> f32 accumulate)
// ---------------------------------------------------------------------------
typedef __attribute__((ext_vector_type(16))) __bf16 v16bf;
typedef __attribute__((ext_vector_type(8)))  __bf16 v8bf;
typedef __attribute__((ext_vector_type(8)))  float  v8f;
typedef __attribute__((ext_vector_type(4)))  float  v4f;
typedef __attribute__((ext_vector_type(4)))  unsigned short v4u16;
typedef __attribute__((ext_vector_type(4)))  unsigned int   v4u;
typedef __attribute__((ext_vector_type(8)))  int            v8i;
typedef __attribute__((ext_vector_type(4)))  int            v4i;

#define B_DIM 8
#define L_DIM 1024
#define E_DIM 1024
#define H_DIM 16
#define A_DIM 64
#define QK_SCALE 0.125f    // 64^-0.5

// Tensor Data Mover availability / arity (ROCm 7.2 = 5-arg; therock = 6-arg).
#if __has_builtin(__builtin_amdgcn_tensor_load_to_lds)
#define HAVE_TDM 1
#if __has_include(<hip/amd_detail/amd_gfx1250_TDM.h>)
#define TDM_SIX_ARGS 1
#endif
#endif

__device__ __forceinline__ void wave_fence() {
#if __has_builtin(__builtin_amdgcn_wave_barrier)
    __builtin_amdgcn_wave_barrier();   // LDS in-order per wave (DScnt) does the rest
#else
    __syncthreads();
#endif
}

// round-to-nearest-even float -> bf16 bits
__device__ __forceinline__ unsigned short f2bf(float f) {
    unsigned u = __float_as_uint(f);
    unsigned r = (u + 0x7FFFu + ((u >> 16) & 1u)) >> 16;
    return (unsigned short)r;
}

__device__ __forceinline__ v8f wmma_bf16(v16bf a, v16bf b, v8f c) {
    return __builtin_amdgcn_wmma_f32_16x16x32_bf16(
        false, a, false, b, (short)0, c, false, false);
}

// 16x32 bf16 A/B fragment from an LDS row-major (stride 64) tile.
// lane holds row (lane&15); K chunks [g*8,g*8+8) and [16+g*8,...), g=lane>>4.
__device__ __forceinline__ v16bf frag_ld(const unsigned short* rowptr,
                                         int ks, int lane) {
    const unsigned short* p = rowptr + ks + ((lane >> 4) << 3);
    union { v16bf v; v8bf h[2]; } u;
    u.h[0] = *(const v8bf*)(p);
    u.h[1] = *(const v8bf*)(p + 16);
    return u.v;
}

// ---------------------------------------------------------------------------
// TDM: 2D bf16 tile load (tile_d0 contiguous elems x tile_d1 rows, row stride
// `stride0` elems) from global into LDS (packed, stride = tile_d0).
// D# built per CDNA5 ISA. Issued by one wave; EXEC ignored.
// ---------------------------------------------------------------------------
#ifdef HAVE_TDM
__device__ __forceinline__ void tdm_load_2d(const void* lds_dst,
                                            const void* gsrc,
                                            unsigned tile_d0, unsigned tile_d1,
                                            unsigned tensor_d0, unsigned tensor_d1,
                                            unsigned stride0) {
    unsigned long long ga = (unsigned long long)gsrc;
    unsigned lds_addr = (unsigned)(unsigned long long)lds_dst;
    v4u g0;
    g0[0] = 1u;                                          // count=1, user mode
    g0[1] = lds_addr;
    g0[2] = (unsigned)(ga & 0xFFFFFFFFu);
    g0[3] = (unsigned)((ga >> 32) & 0x01FFFFFFu) | (2u << 30);  // type=2
    v8i g1;
    g1[0] = (int)(1u << 16);                             // data_size = 2 bytes
    g1[1] = (int)((tensor_d0 & 0xFFFFu) << 16);
    g1[2] = (int)((tensor_d0 >> 16) & 0xFFFFu) |
            (int)((tensor_d1 & 0xFFFFu) << 16);
    g1[3] = (int)((tensor_d1 >> 16) & 0xFFFFu) |
            (int)(tile_d0 << 16);
    g1[4] = (int)(tile_d1 & 0xFFFFu);
    g1[5] = (int)stride0;
    g1[6] = 0;
    g1[7] = 0;
    v4i z = {0, 0, 0, 0};
#ifdef TDM_SIX_ARGS
    v8i z8 = {0, 0, 0, 0, 0, 0, 0, 0};
    __builtin_amdgcn_tensor_load_to_lds(g0, g1, z, z, z8, 0);
#else
    __builtin_amdgcn_tensor_load_to_lds(g0, g1, z, z, 0);
#endif
}
#endif

// Fallback staging: rows x 64 bf16 tile into LDS with 16B vector ops.
__device__ __forceinline__ void stage_bf16(unsigned short* dstLds,
                                           const unsigned short* src,
                                           int rows, size_t srcStride, int t) {
    int chunks = rows * 8;
    for (int c = t; c < chunks; c += 256) {
        int e = c * 8;
        int row = e >> 6, col = e & 63;
        *(uint4*)&dstLds[row * 64 + col] =
            *(const uint4*)&src[(size_t)row * srcStride + col];
    }
}

// ---------------------------------------------------------------------------
// Elementwise fp32 -> bf16 (streaming: NT on both sides, keep L2 for tiles).
// ---------------------------------------------------------------------------
__global__ __launch_bounds__(256) void cvt_bf16_kernel(
    const float* __restrict__ src, unsigned short* __restrict__ dst, int n4)
{
    int i = blockIdx.x * 256 + threadIdx.x;
    if (i < n4) {
        v4f f = __builtin_nontemporal_load((const v4f*)src + i);
        v4u16 h;
        h.x = f2bf(f.x); h.y = f2bf(f.y); h.z = f2bf(f.z); h.w = f2bf(f.w);
        __builtin_nontemporal_store(h, (v4u16*)dst + i);
    }
}

// ---------------------------------------------------------------------------
// bf16 GEMM: Out[m,n] = sum_k X[m,k] * W[n,k]   (X:[M,K], W:[N,K], bf16)
// mode 0: scatter Q (x scale) / K row-major [B,H,L,A]; V transposed [B,H,A,L]
// mode 1: write f32 Out[m*N+n] (non-temporal; final output)
// Block 64x128, 8 waves (4x2). Double-buffered LDS: TDM for tile i+1 issued
// right after the barrier, overlapping compute on tile i; one barrier/tile.
// ---------------------------------------------------------------------------
#define BM 64
#define BN 128
#define BK 64

__global__ __launch_bounds__(256) void gemm_kernel(
    const unsigned short* __restrict__ X, const unsigned short* __restrict__ W,
    float* __restrict__ OutF,
    unsigned short* __restrict__ Qw, unsigned short* __restrict__ Kw,
    unsigned short* __restrict__ Vw,
    int N, int K, int mode)
{
    __shared__ unsigned short As[2][BM * BK];
    __shared__ unsigned short Bs[2][BN * BK];

    const int t    = threadIdx.x;
    const int lane = t & 31;
    const int wave = t >> 5;
    const int wm   = wave >> 1;
    const int wn   = wave & 1;
    const int m0   = blockIdx.x * BM;
    const int n0   = blockIdx.y * BN;
    const int nTiles = K / BK;

    // prologue: stage tile 0
#ifdef HAVE_TDM
    if (t < 32) {
        tdm_load_2d(As[0], X + (size_t)m0 * K, BK, BM, K, 8192, K);
        tdm_load_2d(Bs[0], W + (size_t)n0 * K, BK, BN, K, N, K);
    }
#else
    stage_bf16(As[0], X + (size_t)m0 * K, BM, K, t);
    stage_bf16(Bs[0], W + (size_t)n0 * K, BN, K, t);
#endif

    v8f acc[4] = {};

    for (int kt = 0; kt < nTiles; ++kt) {
        const int cur = kt & 1, nxt = cur ^ 1;
#ifdef HAVE_TDM
        if (t < 32) __builtin_amdgcn_s_wait_tensorcnt(0);   // cur pair landed
#endif
        __syncthreads();   // cur visible to all; all done reading buffer nxt

        if (kt + 1 < nTiles) {          // kick off next tile's DMA, then compute
            const int k1 = (kt + 1) * BK;
#ifdef HAVE_TDM
            if (t < 32) {
                tdm_load_2d(As[nxt], X + (size_t)m0 * K + k1, BK, BM, K, 8192, K);
                tdm_load_2d(Bs[nxt], W + (size_t)n0 * K + k1, BK, BN, K, N, K);
            }
#else
            stage_bf16(As[nxt], X + (size_t)m0 * K + k1, BM, K, t);
            stage_bf16(Bs[nxt], W + (size_t)n0 * K + k1, BN, K, t);
#endif
        }

        const unsigned short* Ac = As[cur];
        const unsigned short* Bc = Bs[cur];
#pragma unroll
        for (int ks = 0; ks < BK; ks += 32) {
            v16bf a  = frag_ld(&Ac[(wm * 16 + (lane & 15)) * BK], ks, lane);
            v16bf b0 = frag_ld(&Bc[(wn * 64 +  0 + (lane & 15)) * BK], ks, lane);
            v16bf b1 = frag_ld(&Bc[(wn * 64 + 16 + (lane & 15)) * BK], ks, lane);
            v16bf b2 = frag_ld(&Bc[(wn * 64 + 32 + (lane & 15)) * BK], ks, lane);
            v16bf b3 = frag_ld(&Bc[(wn * 64 + 48 + (lane & 15)) * BK], ks, lane);
            acc[0] = wmma_bf16(a, b0, acc[0]);
            acc[1] = wmma_bf16(a, b1, acc[1]);
            acc[2] = wmma_bf16(a, b2, acc[2]);
            acc[3] = wmma_bf16(a, b3, acc[3]);
        }
    }

    // epilogue: lane holds N=lane&15, rows M = (lane>>4)*8 + r
    const int mbase = m0 + wm * 16 + ((lane >> 4) << 3);
#pragma unroll
    for (int nt = 0; nt < 4; ++nt) {
        int n = n0 + wn * 64 + nt * 16 + (lane & 15);
#pragma unroll
        for (int r = 0; r < 8; ++r) {
            float v = acc[nt][r];
            int m   = mbase + r;
            if (mode) {
                __builtin_nontemporal_store(v, &OutF[(size_t)m * N + n]);
            } else {
                int hh = n / 192, r3 = n % 192;
                int sel = r3 >> 6, a = r3 & 63;
                int b = m >> 10, l = m & 1023;
                if (sel == 0) {
                    Qw[((size_t)(b * H_DIM + hh) * L_DIM + l) * A_DIM + a] =
                        f2bf(v * QK_SCALE);
                } else if (sel == 1) {
                    Kw[((size_t)(b * H_DIM + hh) * L_DIM + l) * A_DIM + a] = f2bf(v);
                } else {
                    // V stored transposed: [B,H,A,L] so P.V B-tiles are 2D loads
                    Vw[((size_t)(b * H_DIM + hh) * A_DIM + a) * L_DIM + l] = f2bf(v);
                }
            }
        }
    }
}

// ---------------------------------------------------------------------------
// Flash attention: per (b,h), 128 q-rows per block, 8 waves x 16 q-rows.
// Double-buffered K/V tiles via TDM; online softmax; bias/mask streamed NT;
// per-wave P turnaround uses a wave fence only (one barrier per tile).
// ---------------------------------------------------------------------------
__global__ __launch_bounds__(256) void attn_kernel(
    const unsigned short* __restrict__ Qw, const unsigned short* __restrict__ Kw,
    const unsigned short* __restrict__ Vw,
    const float* __restrict__ bias, const unsigned char* __restrict__ mask,
    unsigned short* __restrict__ Ob)
{
    __shared__ unsigned short Qs[128 * 64];
    __shared__ unsigned short Ks[2][64 * 64];
    __shared__ unsigned short VsT[2][64 * 64];    // [a][key]
    __shared__ unsigned short Ps[8][16 * 64];     // per-wave P tile

    const int t    = threadIdx.x;
    const int lane = t & 31;
    const int wave = t >> 5;
    const int bh   = blockIdx.y;                  // b*H + h
    const int q0   = blockIdx.x * 128;
    const size_t headOff = (size_t)bh * L_DIM * A_DIM;   // same extent for V^T

    // prologue: Q tile + first K/V tiles (all waited by first s_wait_tensorcnt)
#ifdef HAVE_TDM
    if (t < 32) {
        tdm_load_2d(Qs, Qw + headOff + (size_t)q0 * A_DIM, 64, 128,
                    A_DIM, L_DIM, A_DIM);
        tdm_load_2d(Ks[0],  Kw + headOff, 64, 64, A_DIM, L_DIM, A_DIM);
        tdm_load_2d(VsT[0], Vw + headOff, 64, 64, L_DIM, A_DIM, L_DIM);
    }
#else
    stage_bf16(Qs, Qw + headOff + (size_t)q0 * A_DIM, 128, A_DIM, t);
    stage_bf16(Ks[0],  Kw + headOff, 64, A_DIM, t);
    stage_bf16(VsT[0], Vw + headOff, 64, L_DIM, t);
#endif

    float mrun[8], lrun[8];
#pragma unroll
    for (int r = 0; r < 8; ++r) { mrun[r] = -3.0e38f; lrun[r] = 0.0f; }
    v8f acc[4] = {};

    const int    qrow0    = q0 + wave * 16 + ((lane >> 4) << 3);
    const size_t biasBase = (size_t)bh * L_DIM * L_DIM;

    for (int kt = 0; kt < L_DIM / 64; ++kt) {
        const int cur = kt & 1, nxt = cur ^ 1;
        const int kbase = kt * 64;
#ifdef HAVE_TDM
        if (t < 32) __builtin_amdgcn_s_wait_tensorcnt(0);
#endif
        __syncthreads();   // cur tiles visible; prev tile fully consumed

        if (kt + 1 < L_DIM / 64) {      // overlap next tile's DMA with compute
#ifdef HAVE_TDM
            if (t < 32) {
                tdm_load_2d(Ks[nxt], Kw + headOff + (size_t)(kbase + 64) * A_DIM,
                            64, 64, A_DIM, L_DIM, A_DIM);
                tdm_load_2d(VsT[nxt], Vw + headOff + kbase + 64,
                            64, 64, L_DIM, A_DIM, L_DIM);
            }
#else
            stage_bf16(Ks[nxt], Kw + headOff + (size_t)(kbase + 64) * A_DIM,
                       64, A_DIM, t);
            stage_bf16(VsT[nxt], Vw + headOff + kbase + 64, 64, L_DIM, t);
#endif
            __builtin_prefetch(&bias[biasBase + (size_t)qrow0 * L_DIM + kbase + 64],
                               0, 0);
        }

        // S = (Q*scale) K^T : 4 key tiles of 16, K-dim A=64
        const unsigned short* Kc = Ks[cur];
        v8f s[4] = {};
#pragma unroll
        for (int ks = 0; ks < 64; ks += 32) {
            v16bf a  = frag_ld(&Qs[(wave * 16 + (lane & 15)) * 64], ks, lane);
            v16bf b0 = frag_ld(&Kc[( 0 + (lane & 15)) * 64], ks, lane);
            v16bf b1 = frag_ld(&Kc[(16 + (lane & 15)) * 64], ks, lane);
            v16bf b2 = frag_ld(&Kc[(32 + (lane & 15)) * 64], ks, lane);
            v16bf b3 = frag_ld(&Kc[(48 + (lane & 15)) * 64], ks, lane);
            s[0] = wmma_bf16(a, b0, s[0]);
            s[1] = wmma_bf16(a, b1, s[1]);
            s[2] = wmma_bf16(a, b2, s[2]);
            s[3] = wmma_bf16(a, b3, s[3]);
        }

        // bias + mask: the HBM-bound 640MB stream -> non-temporal loads
#pragma unroll
        for (int nt = 0; nt < 4; ++nt) {
            int kg = kbase + nt * 16 + (lane & 15);
#pragma unroll
            for (int r = 0; r < 8; ++r) {
                size_t bi = biasBase + (size_t)(qrow0 + r) * L_DIM + kg;
                float v = s[nt][r] + __builtin_nontemporal_load(&bias[bi]);
                if (__builtin_nontemporal_load(&mask[bi])) v = -3.0e38f;
                s[nt][r] = v;
            }
        }

        // online softmax (row spans 16 lanes of the half-wave)
#pragma unroll
        for (int r = 0; r < 8; ++r) {
            float mx = fmaxf(fmaxf(s[0][r], s[1][r]), fmaxf(s[2][r], s[3][r]));
            mx = fmaxf(mx, __shfl_xor(mx, 1));
            mx = fmaxf(mx, __shfl_xor(mx, 2));
            mx = fmaxf(mx, __shfl_xor(mx, 4));
            mx = fmaxf(mx, __shfl_xor(mx, 8));
            float mnew = fmaxf(mrun[r], mx);
            float corr = __expf(mrun[r] - mnew);
            float rs = 0.0f;
#pragma unroll
            for (int nt = 0; nt < 4; ++nt) {
                float p = __expf(s[nt][r] - mnew);
                s[nt][r] = p;
                rs += p;
            }
            rs += __shfl_xor(rs, 1);
            rs += __shfl_xor(rs, 2);
            rs += __shfl_xor(rs, 4);
            rs += __shfl_xor(rs, 8);
            lrun[r] = lrun[r] * corr + rs;
            mrun[r] = mnew;
#pragma unroll
            for (int nt = 0; nt < 4; ++nt) acc[nt][r] *= corr;
        }

        // C-layout -> A-layout via per-wave LDS P tile (wave-local: no block
        // barrier needed; LDS is in-order within a wave)
        unsigned short* Pw = Ps[wave];
#pragma unroll
        for (int nt = 0; nt < 4; ++nt) {
#pragma unroll
            for (int r = 0; r < 8; ++r) {
                int Mi = ((lane >> 4) << 3) + r;
                Pw[Mi * 64 + nt * 16 + (lane & 15)] = f2bf(s[nt][r]);
            }
        }
        wave_fence();

        // O += P V  (B-tiles are rows of V^T)
        const unsigned short* Vc = VsT[cur];
#pragma unroll
        for (int ks = 0; ks < 64; ks += 32) {
            v16bf a  = frag_ld(&Pw[(lane & 15) * 64], ks, lane);
            v16bf b0 = frag_ld(&Vc[( 0 + (lane & 15)) * 64], ks, lane);
            v16bf b1 = frag_ld(&Vc[(16 + (lane & 15)) * 64], ks, lane);
            v16bf b2 = frag_ld(&Vc[(32 + (lane & 15)) * 64], ks, lane);
            v16bf b3 = frag_ld(&Vc[(48 + (lane & 15)) * 64], ks, lane);
            acc[0] = wmma_bf16(a, b0, acc[0]);
            acc[1] = wmma_bf16(a, b1, acc[1]);
            acc[2] = wmma_bf16(a, b2, acc[2]);
            acc[3] = wmma_bf16(a, b3, acc[3]);
        }
    }

    // normalize, write bf16 [B, L, E] (head-transposed back; reused by GEMM 3)
    const int b = bh >> 4, h = bh & 15;
#pragma unroll
    for (int nt = 0; nt < 4; ++nt) {
        int a = nt * 16 + (lane & 15);
#pragma unroll
        for (int r = 0; r < 8; ++r) {
            int q = qrow0 + r;
            float v = acc[nt][r] / lrun[r];
            Ob[((size_t)(b * L_DIM + q)) * E_DIM + h * A_DIM + a] = f2bf(v);
        }
    }
}

// ---------------------------------------------------------------------------
// Launch: convert -> QKV GEMM -> flash attention -> output GEMM.
// Workspace (bytes): Xbf 16M | Wqkv 6M | Wout 2M | Q 16M | K 16M | V 16M |
//                    Obf 16M  = 88 MB total.
// ---------------------------------------------------------------------------
extern "C" void kernel_launch(void* const* d_in, const int* in_sizes, int n_in,
                              void* d_out, int out_size, void* d_ws, size_t ws_size,
                              hipStream_t stream) {
    const float*         emb  = (const float*)d_in[0];
    const unsigned char* mask = (const unsigned char*)d_in[1];
    const float*         bias = (const float*)d_in[2];
    const float*         Wqkv = (const float*)d_in[3];
    const float*         Wout = (const float*)d_in[4];
    float*               out  = (float*)d_out;

    char* ws = (char*)d_ws;
    const size_t MB = 1024 * 1024;
    unsigned short* Xbf   = (unsigned short*)(ws);
    unsigned short* Wqkvb = (unsigned short*)(ws + 16 * MB);
    unsigned short* Woutb = (unsigned short*)(ws + 22 * MB);
    unsigned short* Qws   = (unsigned short*)(ws + 24 * MB);
    unsigned short* Kws   = (unsigned short*)(ws + 40 * MB);
    unsigned short* Vws   = (unsigned short*)(ws + 56 * MB);
    unsigned short* Obf   = (unsigned short*)(ws + 72 * MB);

    dim3 blk(256);
    // 0) one-shot bf16 conversions (streaming)
    cvt_bf16_kernel<<<dim3((B_DIM * L_DIM * E_DIM / 4 + 255) / 256), blk, 0, stream>>>(
        emb, Xbf, B_DIM * L_DIM * E_DIM / 4);
    cvt_bf16_kernel<<<dim3((3 * E_DIM * E_DIM / 4 + 255) / 256), blk, 0, stream>>>(
        Wqkv, Wqkvb, 3 * E_DIM * E_DIM / 4);
    cvt_bf16_kernel<<<dim3((E_DIM * E_DIM / 4 + 255) / 256), blk, 0, stream>>>(
        Wout, Woutb, E_DIM * E_DIM / 4);
    // 1) QKV projection: [8192,1024] x [3072,1024]^T (scatter Q/K/V^T)
    gemm_kernel<<<dim3((B_DIM * L_DIM) / BM, (3 * E_DIM) / BN), blk, 0, stream>>>(
        Xbf, Wqkvb, nullptr, Qws, Kws, Vws, 3 * E_DIM, E_DIM, 0);
    // 2) flash attention
    attn_kernel<<<dim3(L_DIM / 128, B_DIM * H_DIM), blk, 0, stream>>>(
        Qws, Kws, Vws, bias, mask, Obf);
    // 3) output projection: [8192,1024] x [1024,1024]^T
    gemm_kernel<<<dim3((B_DIM * L_DIM) / BM, E_DIM / BN), blk, 0, stream>>>(
        Obf, Woutb, out, nullptr, nullptr, nullptr, E_DIM, E_DIM, 1);
}
